// Attention_89111981457444
// MI455X (gfx1250) — matrix operand order repaired and checked by reading, not compiled, not run on hardware
//
#include <hip/hip_runtime.h>
#include <hip/hip_bf16.h>

// ---------------------------------------------------------------------------
// x[4,2048,1024] @ W_in[1024,3072] -> 16-head attention (D=64) -> @ W_out.
// bf16 WMMA (v_wmma_f32_16x16x32_bf16) everywhere, f32 accumulation.
// ---------------------------------------------------------------------------
#define SEQ_L   2048
#define FEAT    1024
#define NHEADS  16
#define HEADD   64
#define MROWS   8192

typedef __attribute__((ext_vector_type(16))) __bf16 v16bf;
typedef __attribute__((ext_vector_type(8)))  float  v8f;

union FragU { v16bf v; unsigned u[8]; };

// --- async global->LDS (CDNA5, ASYNCcnt) with compile-safe probing ---------
#if defined(__has_builtin)
#  if __has_builtin(__builtin_amdgcn_global_load_async_to_lds_b128) && \
      __has_builtin(__builtin_amdgcn_s_wait_asynccnt)
#    define HAVE_ASYNC_LDS 1
#  endif
#endif
#ifndef HAVE_ASYNC_LDS
#  define HAVE_ASYNC_LDS 0
#endif

typedef int v4i __attribute__((__vector_size__(16)));
typedef __attribute__((address_space(1))) v4i* gptr_b128;   // global ptr to b128
typedef __attribute__((address_space(3))) v4i* lptr_b128;   // LDS ptr to b128

__device__ __forceinline__ void copy16_g2l(__bf16* l, const __bf16* g) {
#if HAVE_ASYNC_LDS
  __builtin_amdgcn_global_load_async_to_lds_b128(
      (gptr_b128)(g), (lptr_b128)(l), 0, 0);
#else
  *(uint4*)l = *(const uint4*)g;
#endif
}
__device__ __forceinline__ void wait_async_all() {
#if HAVE_ASYNC_LDS
  __builtin_amdgcn_s_wait_asynccnt(0);
#endif
}

// --- WMMA fragment loaders (wave32; contiguous k = 16h+i, same in A and B) -----
// A (16x32 bf16): row = lane%16; element i: K = (lane/16)*16 + i
__device__ __forceinline__ v16bf lds_load_a(const __bf16* base, int pitch, int lane) {
  const int r = lane & 15, h = lane >> 4;
  const __bf16* rowp = base + r * pitch + h * 16;
  FragU f;
#pragma unroll
  for (int j = 0; j < 8; ++j) {
    f.u[j] = *(const unsigned*)(rowp + 2 * j);
  }
  return f.v;
}
// B (32x16 bf16): col = lane%16; element i: K = (lane/16)*16 + i.
// Tiles stored transposed [n][k] in LDS -> 8 contiguous dword reads.
__device__ __forceinline__ v16bf lds_load_b(const __bf16* base, int pitch, int lane) {
  const int n = lane & 15, h = lane >> 4;
  const __bf16* rowp = base + n * pitch + h * 16;
  FragU f;
#pragma unroll
  for (int j = 0; j < 8; ++j) f.u[j] = *(const unsigned*)(rowp + 2 * j);
  return f.v;
}

__device__ __forceinline__ unsigned pack_bf16x2(float a, float b) {
  union { __bf16 h[2]; unsigned u; } p;
  p.h[0] = (__bf16)a; p.h[1] = (__bf16)b;
  return p.u;
}

#define PA  34                 // A LDS pitch (+2 pad)
#define PBT 36                 // B^T LDS pitch (+4 pad)
#define PQ  72                 // attention pitch: rows 144B -> 16B aligned for B128
#define A_TILE (128 * PA)
#define B_TILE (128 * PBT)
#define Q_TILE (64 * PQ)

// ===========================================================================
// Kernel 1: QKV projection.  Block 256thr/8 waves, tile 128Mx128N, K-step 32,
// double-buffered LDS + register prefetch.  Wave tile 32x64 = 8 WMMA/step.
// Scatters bf16 Q(*0.125)/K/V into [N,H,L,64].
// ===========================================================================
__global__ __launch_bounds__(256) void qkv_proj_kernel(
    const float* __restrict__ x, const float* __restrict__ W_in,
    const float* __restrict__ b_in,
    __bf16* __restrict__ Qb, __bf16* __restrict__ Kb, __bf16* __restrict__ Vb)
{
  __shared__ __bf16 Asm[2 * A_TILE];
  __shared__ __bf16 Bsm[2 * B_TILE];
  const int t = threadIdx.x, lane = t & 31, wave = t >> 5;
  const int m0 = blockIdx.y * 128, n0 = blockIdx.x * 128;
  const int wm = (wave & 3) * 32, wn = (wave >> 2) * 64;
  const int arow = t >> 1, akl = (t & 1) * 16;       // A: 2 thr/row, 16 f32 each
  const int bkr = t >> 3, bnc = (t & 7) * 16;        // B: 32k x 128n, 16 f32 each

  float4 a4[4], b4[4];
  auto load_g = [&](int k0) {
    const float* asrc = x + (size_t)(m0 + arow) * FEAT + k0 + akl;
#pragma unroll
    for (int c = 0; c < 4; ++c) a4[c] = *(const float4*)(asrc + 4 * c);
    const float* bsrc = W_in + (size_t)(k0 + bkr) * 3072 + n0 + bnc;
#pragma unroll
    for (int c = 0; c < 4; ++c) b4[c] = *(const float4*)(bsrc + 4 * c);
  };
  auto store_l = [&](int buf) {
    unsigned* ad = (unsigned*)(Asm + buf * A_TILE + arow * PA + akl);
#pragma unroll
    for (int c = 0; c < 4; ++c) {
      ad[2 * c + 0] = pack_bf16x2(a4[c].x, a4[c].y);
      ad[2 * c + 1] = pack_bf16x2(a4[c].z, a4[c].w);
    }
    __bf16* bd = Bsm + buf * B_TILE + bkr;           // transposed [n][k]
#pragma unroll
    for (int c = 0; c < 4; ++c) {
      float v[4] = {b4[c].x, b4[c].y, b4[c].z, b4[c].w};
#pragma unroll
      for (int q = 0; q < 4; ++q) bd[(bnc + 4 * c + q) * PBT] = (__bf16)v[q];
    }
  };

  v8f acc[2][4] = {};
  load_g(0);
  for (int kt = 0; kt < FEAT / 32; ++kt) {
    const int buf = kt & 1;
    store_l(buf);
    __syncthreads();
    if (kt + 1 < FEAT / 32) load_g((kt + 1) * 32);   // prefetch next K-step
    v16bf af[2], bfr[4];
#pragma unroll
    for (int i = 0; i < 2; ++i)
      af[i] = lds_load_a(Asm + buf * A_TILE + (wm + i * 16) * PA, PA, lane);
#pragma unroll
    for (int j = 0; j < 4; ++j)
      bfr[j] = lds_load_b(Bsm + buf * B_TILE + (wn + j * 16) * PBT, PBT, lane);
#pragma unroll
    for (int i = 0; i < 2; ++i)
#pragma unroll
      for (int j = 0; j < 4; ++j)
        acc[i][j] = __builtin_amdgcn_wmma_f32_16x16x32_bf16(
            false, af[i], false, bfr[j], (short)0, acc[i][j], false, false);
  }

  const int half = lane >> 4, nl = lane & 15;
#pragma unroll
  for (int j = 0; j < 4; ++j) {
    const int colb = n0 + wn + j * 16;               // 16-col tile: same q/k/v+head
    const int which = colb >> 10, hc = (colb & 1023) >> 6;
    __bf16* dst = (which == 0) ? Qb : ((which == 1) ? Kb : Vb);
    const float scl = (which == 0) ? 0.125f : 1.0f;  // fold softmax scale into Q
    const int col = colb + nl, d = col & 63;
    const float bias = b_in[col];
#pragma unroll
    for (int i = 0; i < 2; ++i)
#pragma unroll
      for (int r = 0; r < 8; ++r) {
        const int row = m0 + wm + i * 16 + r + 8 * half;
        const int nb = row >> 11, l = row & (SEQ_L - 1);
        const float v = (acc[i][j][r] + bias) * scl;
        dst[((((size_t)nb * NHEADS + hc) * SEQ_L) + l) * HEADD + d] = (__bf16)v;
      }
  }
}

// ===========================================================================
// Kernel 2: flash attention per (n,h).  128thr/4 waves, 64 q-rows per block,
// key blocks of 64, online softmax.  K tile staged via async global->LDS
// (double buffered); V prefetched through registers and transposed to [d][key].
// ===========================================================================
__global__ __launch_bounds__(128) void attn_kernel(
    const __bf16* __restrict__ Qb, const __bf16* __restrict__ Kb,
    const __bf16* __restrict__ Vb, __bf16* __restrict__ Ob)
{
  __shared__ __bf16 Qs[Q_TILE];           // [qrow][d]
  __shared__ __bf16 Ks[2 * Q_TILE];       // [key][d]  (B-frags for S)
  __shared__ __bf16 Vts[2 * Q_TILE];      // [d][key]  (B-frags for PV)
  __shared__ __bf16 Ps[4 * 16 * PQ];      // per-wave P tile [16][64]

  const int t = threadIdx.x, lane = t & 31, wave = t >> 5;
  const int half = lane >> 4, nl = lane & 15;
  const int q0 = blockIdx.x * 64;
  const int nh = blockIdx.y;
  const size_t base = (size_t)nh * SEQ_L * HEADD;

  uint4 vreg[4];
  auto issue_stage = [&](int kb, int buf) {
#pragma unroll
    for (int e = 0; e < 4; ++e) {
      const int c = t + e * 128, row = c >> 3, dl = (c & 7) * 8;
      copy16_g2l(Ks + buf * Q_TILE + row * PQ + dl,
                 Kb + base + (size_t)(kb + row) * HEADD + dl);
      vreg[e] = *(const uint4*)(Vb + base + (size_t)(kb + row) * HEADD + dl);
    }
  };
  auto scatter_v = [&](int buf) {
#pragma unroll
    for (int e = 0; e < 4; ++e) {
      const int c = t + e * 128, row = c >> 3, dl = (c & 7) * 8;
      unsigned w[4] = {vreg[e].x, vreg[e].y, vreg[e].z, vreg[e].w};
#pragma unroll
      for (int q = 0; q < 8; ++q) {
        union { unsigned u; __bf16 h[2]; } pu; pu.u = w[q >> 1];
        Vts[buf * Q_TILE + (dl + q) * PQ + row] = pu.h[q & 1];
      }
    }
  };

  // stage Q tile (async) + first K/V block
#pragma unroll
  for (int e = 0; e < 4; ++e) {
    const int c = t + e * 128, row = c >> 3, dl = (c & 7) * 8;
    copy16_g2l(Qs + row * PQ + dl, Qb + base + (size_t)(q0 + row) * HEADD + dl);
  }
  issue_stage(0, 0);
  wait_async_all();
  __syncthreads();

  v16bf aq[2];
  aq[0] = lds_load_a(Qs + wave * 16 * PQ,      PQ, lane);
  aq[1] = lds_load_a(Qs + wave * 16 * PQ + 32, PQ, lane);

  v8f o[4] = {};
  float mrow[8], lrow[8];
#pragma unroll
  for (int r = 0; r < 8; ++r) { mrow[r] = -1e30f; lrow[r] = 0.f; }
  __bf16* Pw = Ps + wave * 16 * PQ;

  const int NT = SEQ_L / 64;
  for (int kbi = 0; kbi < NT; ++kbi) {
    const int buf = kbi & 1;
    scatter_v(buf);                        // V for this block (prefetched regs)
    wait_async_all();                      // K async writes for this block done
    __syncthreads();
    if (kbi + 1 < NT) issue_stage((kbi + 1) * 64, buf ^ 1);  // prefetch next

    // S = Q K^T (scale folded into Q)
    v8f s[4] = {};
#pragma unroll
    for (int nt = 0; nt < 4; ++nt)
#pragma unroll
      for (int kk = 0; kk < 2; ++kk) {
        v16bf bk = lds_load_b(Ks + buf * Q_TILE + (nt * 16) * PQ + kk * 32, PQ, lane);
        s[nt] = __builtin_amdgcn_wmma_f32_16x16x32_bf16(
            false, aq[kk], false, bk, (short)0, s[nt], false, false);
      }

    // online softmax: row j lives across a 16-lane group in the C layout
#pragma unroll
    for (int r = 0; r < 8; ++r) {
      float pm = fmaxf(fmaxf(s[0][r], s[1][r]), fmaxf(s[2][r], s[3][r]));
#pragma unroll
      for (int msk = 1; msk < 16; msk <<= 1) pm = fmaxf(pm, __shfl_xor(pm, msk, 32));
      const float mn = fmaxf(mrow[r], pm);
      const float corr = __expf(mrow[r] - mn);
      mrow[r] = mn;
      float ps = 0.f;
#pragma unroll
      for (int nt = 0; nt < 4; ++nt) {
        const float p = __expf(s[nt][r] - mn);
        ps += p;
        Pw[(r + 8 * half) * PQ + nt * 16 + nl] = (__bf16)p;
      }
#pragma unroll
      for (int msk = 1; msk < 16; msk <<= 1) ps += __shfl_xor(ps, msk, 32);
      lrow[r] = lrow[r] * corr + ps;
#pragma unroll
      for (int dt = 0; dt < 4; ++dt) o[dt][r] *= corr;
    }
    __syncthreads();                       // publish P before frag reload

    // O += P @ V
#pragma unroll
    for (int kk = 0; kk < 2; ++kk) {
      v16bf ap = lds_load_a(Pw + kk * 32, PQ, lane);
#pragma unroll
      for (int dt = 0; dt < 4; ++dt) {
        v16bf bv = lds_load_b(Vts + buf * Q_TILE + (dt * 16) * PQ + kk * 32, PQ, lane);
        o[dt] = __builtin_amdgcn_wmma_f32_16x16x32_bf16(
            false, ap, false, bv, (short)0, o[dt], false, false);
      }
    }
  }

  // normalize + store to [N, L, H*64] bf16 for the out-projection
  const int nb = nh >> 4, h = nh & 15;
#pragma unroll
  for (int dt = 0; dt < 4; ++dt)
#pragma unroll
    for (int r = 0; r < 8; ++r) {
      const int row = q0 + wave * 16 + r + 8 * half;
      const int d = dt * 16 + nl;
      Ob[(((size_t)nb * SEQ_L) + row) * FEAT + h * HEADD + d] =
          (__bf16)(o[dt][r] / lrow[r]);
    }
}

// ===========================================================================
// Kernel 3: output projection (A already bf16).  Same pipelined tiling.
// ===========================================================================
__global__ __launch_bounds__(256) void out_proj_kernel(
    const __bf16* __restrict__ Ain, const float* __restrict__ W_out,
    const float* __restrict__ b_out, float* __restrict__ out)
{
  __shared__ __bf16 Asm[2 * A_TILE];
  __shared__ __bf16 Bsm[2 * B_TILE];
  const int t = threadIdx.x, lane = t & 31, wave = t >> 5;
  const int m0 = blockIdx.y * 128, n0 = blockIdx.x * 128;
  const int wm = (wave & 3) * 32, wn = (wave >> 2) * 64;
  const int arow = t >> 1, akl = (t & 1) * 16;
  const int bkr = t >> 3, bnc = (t & 7) * 16;

  uint4 a8[2];
  float4 b4[4];
  auto load_g = [&](int k0) {
    const uint4* asrc = (const uint4*)(Ain + (size_t)(m0 + arow) * FEAT + k0 + akl);
    a8[0] = asrc[0]; a8[1] = asrc[1];
    const float* bsrc = W_out + (size_t)(k0 + bkr) * FEAT + n0 + bnc;
#pragma unroll
    for (int c = 0; c < 4; ++c) b4[c] = *(const float4*)(bsrc + 4 * c);
  };
  auto store_l = [&](int buf) {
    unsigned* ad = (unsigned*)(Asm + buf * A_TILE + arow * PA + akl);
    ad[0] = a8[0].x; ad[1] = a8[0].y; ad[2] = a8[0].z; ad[3] = a8[0].w;
    ad[4] = a8[1].x; ad[5] = a8[1].y; ad[6] = a8[1].z; ad[7] = a8[1].w;
    __bf16* bd = Bsm + buf * B_TILE + bkr;
#pragma unroll
    for (int c = 0; c < 4; ++c) {
      float v[4] = {b4[c].x, b4[c].y, b4[c].z, b4[c].w};
#pragma unroll
      for (int q = 0; q < 4; ++q) bd[(bnc + 4 * c + q) * PBT] = (__bf16)v[q];
    }
  };

  v8f acc[2][4] = {};
  load_g(0);
  for (int kt = 0; kt < FEAT / 32; ++kt) {
    const int buf = kt & 1;
    store_l(buf);
    __syncthreads();
    if (kt + 1 < FEAT / 32) load_g((kt + 1) * 32);
    v16bf af[2], bfr[4];
#pragma unroll
    for (int i = 0; i < 2; ++i)
      af[i] = lds_load_a(Asm + buf * A_TILE + (wm + i * 16) * PA, PA, lane);
#pragma unroll
    for (int j = 0; j < 4; ++j)
      bfr[j] = lds_load_b(Bsm + buf * B_TILE + (wn + j * 16) * PBT, PBT, lane);
#pragma unroll
    for (int i = 0; i < 2; ++i)
#pragma unroll
      for (int j = 0; j < 4; ++j)
        acc[i][j] = __builtin_amdgcn_wmma_f32_16x16x32_bf16(
            false, af[i], false, bfr[j], (short)0, acc[i][j], false, false);
  }

  const int half = lane >> 4, nl = lane & 15;
#pragma unroll
  for (int i = 0; i < 2; ++i)
#pragma unroll
    for (int j = 0; j < 4; ++j)
#pragma unroll
      for (int r = 0; r < 8; ++r) {
        const int row = m0 + wm + i * 16 + r + 8 * half;
        const int col = n0 + wn + j * 16 + nl;
        out[(size_t)row * FEAT + col] = acc[i][j][r] + b_out[col];
      }
}

// ===========================================================================
// Workspace (bf16): Q | K | V each [4,16,2048,64], then Ob [4,2048,1024].
// Total 64 MB of d_ws.
// ===========================================================================
extern "C" void kernel_launch(void* const* d_in, const int* in_sizes, int n_in,
                              void* d_out, int out_size, void* d_ws, size_t ws_size,
                              hipStream_t stream) {
  const float* x     = (const float*)d_in[0];
  const float* W_in  = (const float*)d_in[1];
  const float* b_in  = (const float*)d_in[2];
  const float* W_out = (const float*)d_in[3];
  const float* b_out = (const float*)d_in[4];
  float* out = (float*)d_out;

  const size_t TEN = (size_t)4 * NHEADS * SEQ_L * HEADD;   // 8,388,608 elems
  __bf16* Qb = (__bf16*)d_ws;
  __bf16* Kb = Qb + TEN;
  __bf16* Vb = Kb + TEN;
  __bf16* Ob = Vb + TEN;

  qkv_proj_kernel<<<dim3(3072 / 128, MROWS / 128), 256, 0, stream>>>(x, W_in, b_in, Qb, Kb, Vb);
  attn_kernel<<<dim3(SEQ_L / 64, 4 * NHEADS), 128, 0, stream>>>(Qb, Kb, Vb, Ob);
  out_proj_kernel<<<dim3(FEAT / 128, MROWS / 128), 256, 0, stream>>>(Ob, W_out, b_out, out);
}
